// JambaMambaMixer_9156870275174
// MI455X (gfx1250) — compile-verified, matrix operand order checked by reading, not run on hardware
//
#include <hip/hip_runtime.h>

// ---------------------------------------------------------------------------
// Problem constants (from the reference)
// ---------------------------------------------------------------------------
#define H_DIM  4096
#define DIN    8192
#define N_ST   16
#define K_CONV 4
#define R_DT   256
#define B_SZ   2
#define L_SEQ  2048
#define M_ROWS (B_SZ * L_SEQ)     // 4096 token rows
#define E_DIM  (R_DT + 2 * N_ST)  // 288

typedef __attribute__((ext_vector_type(16))) __bf16 v16bf;
typedef __attribute__((ext_vector_type(8)))  float  v8f;
typedef unsigned short u16;
typedef int v4i_gcc __attribute__((vector_size(16)));   // pointee type expected
                                                        // by the async builtin

union FragBF { v16bf v; unsigned int u[8]; };

#if defined(__has_builtin)
#if __has_builtin(__builtin_amdgcn_global_load_async_to_lds_b128)
#define HAVE_ASYNC_LDS 1
#endif
#endif

static __device__ __forceinline__ u16 f32_to_bf16(float f) {
    unsigned int u = __builtin_bit_cast(unsigned int, f);
    u += 0x7FFFu + ((u >> 16) & 1u);     // round-to-nearest-even
    return (u16)(u >> 16);
}

static __device__ __forceinline__ unsigned int pack_bf16(float lo, float hi) {
    return (unsigned int)f32_to_bf16(lo) | ((unsigned int)f32_to_bf16(hi) << 16);
}

static __device__ __forceinline__ float silu(float x) {
    return x * (1.0f / (1.0f + __expf(-x)));
}

static __device__ __forceinline__ void async_wait0() {
#if defined(__has_builtin)
#if __has_builtin(__builtin_amdgcn_s_wait_asynccnt)
    __builtin_amdgcn_s_wait_asynccnt(0);
    return;
#endif
#endif
    asm volatile("s_wait_asynccnt 0x0" ::: "memory");
}

// copy 16 bytes global -> LDS (async path if available)
static __device__ __forceinline__ void copy16_g2l(const u16* g, u16* l) {
#ifdef HAVE_ASYNC_LDS
    __builtin_amdgcn_global_load_async_to_lds_b128(
        (v4i_gcc*)(void*)g, (v4i_gcc*)(void*)l, 0, 0);
#else
    *(uint4*)l = *(const uint4*)g;
#endif
}

// ---------------------------------------------------------------------------
// bf16 WMMA GEMM:  C[M,N] (fp32) = Abf[M,K] * Bbf[N,K]^T     (both bf16,
// row-major, K contiguous).  128x128 block tile, 256 threads = 8 waves in a
// 4(M) x 2(N) grid; each wave owns 32x64 = 2x4 WMMA tiles (8 accumulators).
// K-step 32 -> v_wmma_f32_16x16x32_bf16.
// Double-buffered LDS tiles filled with global_load_async_to_lds_b128: one
// barrier per K-step, next tile's async fill overlaps current tile's WMMAs.
// OOB columns (N=288 case) are handled by CLAMPING the global address (no
// divergence in the copy loop); the epilogue store is guarded.
// Requires M%128==0, K%32==0.
// ---------------------------------------------------------------------------
__global__ __launch_bounds__(256) void wmma_gemm_bf16(
    const u16* __restrict__ Abf, const u16* __restrict__ Bbf,
    float* __restrict__ C, int M, int N, int K)
{
    __shared__ u16 As[2][128 * 32];   // [buf][row][k]   2 x 8 KB
    __shared__ u16 Bs[2][128 * 32];   // [buf][col][k]   2 x 8 KB

    const int tid    = threadIdx.x;
    const int lane   = tid & 31;
    const int wid    = tid >> 5;
    const int wave_m = wid & 3;    // 4 M-tiles of 32 rows
    const int wave_n = wid >> 2;   // 2 N-strips of 64 cols
    const int block_m = blockIdx.y * 128;
    const int block_n = blockIdx.x * 128;

    v8f acc[2][4] = {};

    // per-thread copy coordinates: 512 16-byte chunks per matrix, 2/thread
    const int rowc[2] = { (0 * 256 + tid) >> 2, (1 * 256 + tid) >> 2 };
    const int koc [2] = { ((0 * 256 + tid) & 3) * 8, ((1 * 256 + tid) & 3) * 8 };

    // issue async fills of tile at k0 into buffer p
    auto issue_tile = [&](int k0, int p) {
        #pragma unroll
        for (int r = 0; r < 2; ++r) {
            int row = rowc[r], ko = koc[r];
            const u16* gA = Abf + (size_t)(block_m + row) * K + (k0 + ko);
            copy16_g2l(gA, &As[p][row * 32 + ko]);
            int colc = block_n + row;
            colc = (colc < N) ? colc : (N - 1);          // clamp, no divergence
            const u16* gB = Bbf + (size_t)colc * K + (k0 + ko);
            copy16_g2l(gB, &Bs[p][row * 32 + ko]);
        }
    };

    issue_tile(0, 0);
    int p = 0;

    for (int k0 = 0; k0 < K; k0 += 32) {
#ifdef HAVE_ASYNC_LDS
        async_wait0();                 // this buffer's fill complete (per wave)
#endif
        __syncthreads();               // ... and visible to all waves

        if (k0 + 32 < K) {
            issue_tile(k0 + 32, p ^ 1);               // overlap with compute
            #pragma unroll
            for (int r = 0; r < 2; ++r)               // prefetch K+64 into L2
                __builtin_prefetch(Abf + (size_t)(block_m + rowc[r]) * K +
                                   (k0 + 64 + koc[r]), 0, 1);
        }

        // ---- fragments (ISA 7.12.2 16-bit layout, packed K-pairs) ----
        const unsigned int* As32 = (const unsigned int*)As[p];
        const unsigned int* Bs32 = (const unsigned int*)Bs[p];
        FragBF fa[2], fb[4];
        const int half = lane >> 4;
        #pragma unroll
        for (int j = 0; j < 8; ++j) {
            int kd = (((j >> 2) << 4) + (half << 3) + ((j & 3) << 1)) >> 1;
            #pragma unroll
            for (int mf = 0; mf < 2; ++mf)
                fa[mf].u[j] = As32[(wave_m * 32 + mf * 16 + (lane & 15)) * 16 + kd];
            #pragma unroll
            for (int nf = 0; nf < 4; ++nf)
                fb[nf].u[j] = Bs32[(wave_n * 64 + nf * 16 + (lane & 15)) * 16 + kd];
        }
        #pragma unroll
        for (int mf = 0; mf < 2; ++mf)
            #pragma unroll
            for (int nf = 0; nf < 4; ++nf)
                acc[mf][nf] = __builtin_amdgcn_wmma_f32_16x16x32_bf16(
                    false, fa[mf].v, false, fb[nf].v, (short)0,
                    acc[mf][nf], false, false);
        p ^= 1;
    }

    // ---- store: VGPR v -> row = v + 8*(lane>=16); col = lane%16 ----
    #pragma unroll
    for (int mf = 0; mf < 2; ++mf) {
        int rbase = block_m + wave_m * 32 + mf * 16 + ((lane >> 4) << 3);
        #pragma unroll
        for (int nf = 0; nf < 4; ++nf) {
            int col = block_n + wave_n * 64 + nf * 16 + (lane & 15);
            if (col < N) {
                #pragma unroll
                for (int v = 0; v < 8; ++v)
                    C[(size_t)(rbase + v) * N + col] = acc[mf][nf][v];
            }
        }
    }
}

// ---------------------------------------------------------------------------
// fp32 -> packed bf16, 4 elements/thread (n % 4 == 0 for all uses)
// ---------------------------------------------------------------------------
__global__ __launch_bounds__(256) void cvt_bf16_kernel(
    const float* __restrict__ in, u16* __restrict__ out, long n)
{
    long i = ((long)blockIdx.x * 256 + threadIdx.x) * 4;
    if (i < n) {
        float4 f = *(const float4*)(in + i);
        unsigned int p0 = pack_bf16(f.x, f.y);
        unsigned int p1 = pack_bf16(f.z, f.w);
        uint2 o; o.x = p0; o.y = p1;
        *(uint2*)(out + i) = o;
    }
}

// ---------------------------------------------------------------------------
// fp32 [R,C] -> bf16 [C,R]  (tiled 32x32 transpose via LDS; R,C % 32 == 0)
// ---------------------------------------------------------------------------
__global__ __launch_bounds__(256) void transpose_cvt_bf16_kernel(
    const float* __restrict__ in, u16* __restrict__ out, int R, int C)
{
    __shared__ float t[32][33];
    const int c0 = blockIdx.x * 32, r0 = blockIdx.y * 32;
    const int tx = threadIdx.x & 31, ty = threadIdx.x >> 5;   // ty: 0..7
    #pragma unroll
    for (int i = 0; i < 32; i += 8)
        t[ty + i][tx] = in[(size_t)(r0 + ty + i) * C + (c0 + tx)];
    __syncthreads();
    #pragma unroll
    for (int i = 0; i < 32; i += 8)
        out[(size_t)(c0 + ty + i) * R + (r0 + tx)] = f32_to_bf16(t[tx][ty + i]);
}

// ---------------------------------------------------------------------------
// Causal depthwise conv (K=4) + bias + SiLU -> xc fp32 (for scan) + bf16 (GEMM A)
// ---------------------------------------------------------------------------
__global__ __launch_bounds__(256) void conv_silu_kernel(
    const float* __restrict__ proj, const float* __restrict__ cw,
    const float* __restrict__ cb, float* __restrict__ xc_f,
    u16* __restrict__ xc_bf)
{
    size_t idx = (size_t)blockIdx.x * 256 + threadIdx.x;
    int d     = (int)(idx % DIN);
    size_t bl = idx / DIN;           // b*L + l
    int l     = (int)(bl % L_SEQ);

    float acc = cb[d];
    #pragma unroll
    for (int k = 0; k < K_CONV; ++k) {
        int ls = l - (K_CONV - 1) + k;
        if (ls >= 0)
            acc += cw[d * K_CONV + k] *
                   proj[(bl + (size_t)(ls - l)) * (size_t)(2 * DIN) + d];
    }
    acc = silu(acc);
    xc_f[bl * (size_t)DIN + d]  = acc;
    xc_bf[bl * (size_t)DIN + d] = f32_to_bf16(acc);
}

// ---------------------------------------------------------------------------
// Split RMSNorm of ssm_p row (288 = 256 dt | 16 B | 16 C); dt_r emitted bf16.
// ---------------------------------------------------------------------------
__global__ __launch_bounds__(256) void rmsnorm_split_kernel(
    const float* __restrict__ ssm_p,
    const float* __restrict__ dt_ln_w, const float* __restrict__ B_ln_w,
    const float* __restrict__ C_ln_w,
    u16* __restrict__ dtr_bf, float* __restrict__ Bn, float* __restrict__ Cn)
{
    const int row = blockIdx.x;
    const int tid = threadIdx.x;
    __shared__ float red[256];
    __shared__ float sscale;

    float v = ssm_p[(size_t)row * E_DIM + tid];
    red[tid] = v * v;
    __syncthreads();
    for (int s = 128; s > 0; s >>= 1) {
        if (tid < s) red[tid] += red[tid + s];
        __syncthreads();
    }
    if (tid == 0) sscale = rsqrtf(red[0] / (float)R_DT + 1e-6f);
    __syncthreads();
    dtr_bf[(size_t)row * R_DT + tid] = f32_to_bf16(v * sscale * dt_ln_w[tid]);

    if (tid < 32) {                          // lanes 0-15: B ; 16-31: C
        int n   = tid & 15;
        int isC = tid >> 4;
        float w = ssm_p[(size_t)row * E_DIM + R_DT + tid];
        float ss = w * w;
        ss += __shfl_xor(ss, 1);
        ss += __shfl_xor(ss, 2);
        ss += __shfl_xor(ss, 4);
        ss += __shfl_xor(ss, 8);             // 16-lane half-sums
        float sc = rsqrtf(ss / (float)N_ST + 1e-6f);
        float lw = isC ? C_ln_w[n] : B_ln_w[n];
        float o  = w * sc * lw;
        if (isC) Cn[(size_t)row * N_ST + n] = o;
        else     Bn[(size_t)row * N_ST + n] = o;
    }
}

// ---------------------------------------------------------------------------
// Selective scan: one lane per (b,d), 16-state recurrence in VGPRs over L.
// B/C staged through LDS in 64-step chunks.  Fuses softplus, D-skip, silu(z)
// gate; emits y directly as bf16 (GEMM-A layout (b,l,d)).
// ---------------------------------------------------------------------------
__global__ __launch_bounds__(256) void scan_kernel(
    const float* __restrict__ xc, const float* __restrict__ dtp,
    const float* __restrict__ dt_bias, const float* __restrict__ A_log,
    const float* __restrict__ Dp, const float* __restrict__ Bn,
    const float* __restrict__ Cn, const float* __restrict__ proj,
    u16* __restrict__ yb_bf)
{
    const int tid  = threadIdx.x;
    const int dblk = blockIdx.x % (DIN / 256);
    const int b    = blockIdx.x / (DIN / 256);
    const int d    = dblk * 256 + tid;

    __shared__ float sB[64 * N_ST];
    __shared__ float sC[64 * N_ST];

    float Aa[N_ST], h[N_ST];
    #pragma unroll
    for (int n = 0; n < N_ST; ++n) {
        Aa[n] = -__expf(A_log[(size_t)d * N_ST + n]);
        h[n]  = 0.0f;
    }
    const float Dv   = Dp[d];
    const float bias = dt_bias[d];

    for (int t0 = 0; t0 < L_SEQ; t0 += 64) {
        __syncthreads();
        for (int i = tid; i < 64 * N_ST; i += 256) {
            int t = i >> 4, n = i & 15;
            size_t g = ((size_t)b * L_SEQ + t0 + t) * N_ST + n;
            sB[i] = Bn[g];
            sC[i] = Cn[g];
        }
        __syncthreads();
        for (int tt = 0; tt < 64; ++tt) {
            const size_t bl = (size_t)b * L_SEQ + (t0 + tt);
            float x  = xc[bl * DIN + d];
            float dv = dtp[bl * DIN + d] + bias;
            float dt = (dv > 20.0f) ? dv : __logf(1.0f + __expf(dv)); // softplus
            float y  = 0.0f;
            #pragma unroll
            for (int n = 0; n < N_ST; ++n) {
                float dA = __expf(dt * Aa[n]);
                h[n] = dA * h[n] + (dt * sB[tt * N_ST + n]) * x;
                y   += h[n] * sC[tt * N_ST + n];
            }
            float z = proj[bl * (size_t)(2 * DIN) + DIN + d];
            yb_bf[bl * DIN + d] = f32_to_bf16((y + Dv * x) * silu(z));
        }
    }
}

// ---------------------------------------------------------------------------
// Host-side pipeline
// ---------------------------------------------------------------------------
extern "C" void kernel_launch(void* const* d_in, const int* in_sizes, int n_in,
                              void* d_out, int out_size, void* d_ws, size_t ws_size,
                              hipStream_t stream) {
    (void)in_sizes; (void)n_in; (void)out_size; (void)ws_size;

    const float* hs        = (const float*)d_in[0];   // (B,L,H)
    const float* in_proj_w = (const float*)d_in[1];   // (H, 2*DIN)
    const float* conv_w    = (const float*)d_in[2];   // (DIN, K)
    const float* conv_b    = (const float*)d_in[3];   // (DIN,)
    const float* x_proj_w  = (const float*)d_in[4];   // (E, DIN)
    const float* dt_proj_w = (const float*)d_in[5];   // (DIN, R)
    const float* dt_bias   = (const float*)d_in[6];   // (DIN,)
    const float* A_log     = (const float*)d_in[7];   // (DIN, N)
    const float* D_param   = (const float*)d_in[8];   // (DIN,)
    const float* out_proj_w= (const float*)d_in[9];   // (H, DIN)
    const float* dt_ln_w   = (const float*)d_in[10];  // (R,)
    const float* B_ln_w    = (const float*)d_in[11];  // (N,)
    const float* C_ln_w    = (const float*)d_in[12];  // (N,)
    float* out = (float*)d_out;                       // (B,L,H)

    // ---- workspace carve-out (256-byte aligned blocks) ----
    char*  base = (char*)d_ws;
    size_t off  = 0;
    auto alloc = [&](size_t bytes) -> void* {
        void* p = base + off;
        off += (bytes + 255) & ~(size_t)255;
        return p;
    };
    float* proj    = (float*)alloc((size_t)M_ROWS * 2 * DIN * 4); // (b,l,2DIN)
    u16*   hs_bf   = (u16*)  alloc((size_t)M_ROWS * H_DIM * 2);
    u16*   ipwT_bf = (u16*)  alloc((size_t)2 * DIN * H_DIM * 2);  // [2DIN, H]
    u16*   xpw_bf  = (u16*)  alloc((size_t)E_DIM * DIN * 2);      // [288, 8192]
    u16*   dtpw_bf = (u16*)  alloc((size_t)DIN * R_DT * 2);       // [8192, 256]
    u16*   opw_bf  = (u16*)  alloc((size_t)H_DIM * DIN * 2);      // [4096, 8192]
    float* xc_f    = (float*)alloc((size_t)M_ROWS * DIN * 4);
    u16*   xc_bf   = (u16*)  alloc((size_t)M_ROWS * DIN * 2);
    float* ssm     = (float*)alloc((size_t)M_ROWS * E_DIM * 4);
    u16*   dtr_bf  = (u16*)  alloc((size_t)M_ROWS * R_DT * 2);
    float* Bn      = (float*)alloc((size_t)M_ROWS * N_ST * 4);
    float* Cn      = (float*)alloc((size_t)M_ROWS * N_ST * 4);
    float* dtp     = (float*)alloc((size_t)M_ROWS * DIN * 4);
    u16*   yb_bf   = (u16*)  alloc((size_t)M_ROWS * DIN * 2);

    // ---- 0) one-time bf16 packing of activations + weights ----
    {
        long n = (long)M_ROWS * H_DIM;
        cvt_bf16_kernel<<<(unsigned)(n / (4 * 256)), 256, 0, stream>>>(hs, hs_bf, n);
    }
    transpose_cvt_bf16_kernel<<<dim3(2 * DIN / 32, H_DIM / 32), 256, 0, stream>>>(
        in_proj_w, ipwT_bf, H_DIM, 2 * DIN);
    {
        long n = (long)E_DIM * DIN;
        cvt_bf16_kernel<<<(unsigned)(n / (4 * 256)), 256, 0, stream>>>(x_proj_w, xpw_bf, n);
    }
    {
        long n = (long)DIN * R_DT;
        cvt_bf16_kernel<<<(unsigned)(n / (4 * 256)), 256, 0, stream>>>(dt_proj_w, dtpw_bf, n);
    }
    {
        long n = (long)H_DIM * DIN;
        cvt_bf16_kernel<<<(unsigned)(n / (4 * 256)), 256, 0, stream>>>(out_proj_w, opw_bf, n);
    }

    // ---- 1) proj = hs @ in_proj_w   (B as [2DIN, H] bf16) ----
    wmma_gemm_bf16<<<dim3(2 * DIN / 128, M_ROWS / 128), 256, 0, stream>>>(
        hs_bf, ipwT_bf, proj, M_ROWS, 2 * DIN, H_DIM);

    // ---- 2) conv + SiLU ----
    conv_silu_kernel<<<(unsigned)((size_t)M_ROWS * DIN / 256), 256, 0, stream>>>(
        proj, conv_w, conv_b, xc_f, xc_bf);

    // ---- 3) ssm_p = xc @ x_proj_w^T ----
    wmma_gemm_bf16<<<dim3((E_DIM + 127) / 128, M_ROWS / 128), 256, 0, stream>>>(
        xc_bf, xpw_bf, ssm, M_ROWS, E_DIM, DIN);

    // ---- 4) split RMSNorm ----
    rmsnorm_split_kernel<<<M_ROWS, 256, 0, stream>>>(
        ssm, dt_ln_w, B_ln_w, C_ln_w, dtr_bf, Bn, Cn);

    // ---- 5) dt_pre = dt_r @ dt_proj_w^T ----
    wmma_gemm_bf16<<<dim3(DIN / 128, M_ROWS / 128), 256, 0, stream>>>(
        dtr_bf, dtpw_bf, dtp, M_ROWS, DIN, R_DT);

    // ---- 6) selective scan + gate ----
    scan_kernel<<<B_SZ * (DIN / 256), 256, 0, stream>>>(
        xc_f, dtp, dt_bias, A_log, D_param, Bn, Cn, proj, yb_bf);

    // ---- 7) out = y @ out_proj_w^T ----
    wmma_gemm_bf16<<<dim3(H_DIM / 128, M_ROWS / 128), 256, 0, stream>>>(
        yb_bf, opw_bf, out, M_ROWS, H_DIM, DIN);
}
